// SSD_10505490006128
// MI455X (gfx1250) — matrix operand order repaired and checked by reading, not compiled, not run on hardware
//
#include <hip/hip_runtime.h>
#include <cstdint>
#include <cmath>

// Problem constants (match reference)
#define B__    2
#define L__    4096
#define DM__   1024
#define DS__   64
#define HD__   32
#define NH__   32
#define CS__   256
#define NC__   (L__ / CS__)              // 16
#define DPROJ__ (DM__ + 2*DS__ + NH__)   // 1184
#define XSTR__  1280                     // padded xbcdt row stride (multiple of 128)

typedef __attribute__((ext_vector_type(16))) __bf16 v16bf;
typedef __attribute__((ext_vector_type(8)))  float  v8f;

__device__ __forceinline__ unsigned short f32_to_bf16(float f) {
  unsigned u = __float_as_uint(f);
  u = u + 0x7FFFu + ((u >> 16) & 1u);   // round-to-nearest-even
  return (unsigned short)(u >> 16);
}
__device__ __forceinline__ float bf16_to_f32(unsigned short u) {
  return __uint_as_float((unsigned)u << 16);
}

// ---------------- fp32 -> bf16 cast ----------------
__global__ void cast_f32_bf16(const float* __restrict__ in,
                              unsigned short* __restrict__ out, int n) {
  int i = blockIdx.x * blockDim.x + threadIdx.x;
  if (i < n) out[i] = f32_to_bf16(in[i]);
}

// ---------------- CDNA5 async global->LDS copy (16B per lane, ASYNCcnt) -------------
__device__ __forceinline__ void async_copy16(unsigned lds_off, const void* gaddr) {
  asm volatile("global_load_async_to_lds_b128 %0, %1, off"
               :: "v"(lds_off), "v"(gaddr) : "memory");
}
__device__ __forceinline__ void wait_async_le4() {
  asm volatile("s_wait_asynccnt 4" ::: "memory");
}
__device__ __forceinline__ void wait_async_0() {
  asm volatile("s_wait_asynccnt 0" ::: "memory");
}

// ---------------- blocked WMMA bf16 GEMM: C[M,N] = A[M,K] @ B[N,K]^T ----------------
// Block = 256 threads (8 waves). Block tile 128x128, K-step 32, double-buffered LDS
// filled with GLOBAL_LOAD_ASYNC_TO_LDS_B128. Wave (wm,wn) computes a 64x32 sub-tile
// as 4x2 WMMA 16x16 accumulators. Requires M%128==0, N%128==0, K%32==0.
#define TSTRIDE 56        // padded LDS row stride in bf16 elements (112 bytes)

__global__ void gemm_bf16_wmma(const unsigned short* __restrict__ A,
                               const unsigned short* __restrict__ B,
                               float* __restrict__ C,
                               int M, int N, int K) {
  __shared__ unsigned short sAB[2][2][128 * TSTRIDE];   // [buf][A|B][...]  57,344 B

  const int t    = threadIdx.x;
  const int lane = t & 31;
  const int w    = t >> 5;          // wave 0..7
  const int wm   = w >> 2;          // 0..1  -> 64-row slab
  const int wn   = w & 3;           // 0..3  -> 32-col slab
  const int half = lane >> 4;       // 0|1
  const int l16  = lane & 15;
  const int kbA  = half << 3;       // A frag K base {0,8}
  const int kbB  = half << 3;       // B frag K base {0,8}

  const int tn = blockIdx.x;        // N/128 tiles
  const int tm = blockIdx.y;        // M/128 tiles
  const int rowA = tm << 7;
  const int colB = tn << 7;

  const int row0 = t >> 2, ch0 = t & 3;
  const int row1 = (t + 256) >> 2, ch1 = ch0;

  const unsigned ldsA0 = (unsigned)(uintptr_t)&sAB[0][0][0];
  const unsigned ldsB0 = (unsigned)(uintptr_t)&sAB[0][1][0];
  const unsigned ldsA1 = (unsigned)(uintptr_t)&sAB[1][0][0];
  const unsigned ldsB1 = (unsigned)(uintptr_t)&sAB[1][1][0];

  auto issue_stage = [&](int k0, unsigned lA, unsigned lB) {
    async_copy16(lA + row0 * 112 + ch0 * 16,
                 (const char*)(A + (size_t)(rowA + row0) * K + k0) + ch0 * 16);
    async_copy16(lB + row0 * 112 + ch0 * 16,
                 (const char*)(B + (size_t)(colB + row0) * K + k0) + ch0 * 16);
    async_copy16(lA + row1 * 112 + ch1 * 16,
                 (const char*)(A + (size_t)(rowA + row1) * K + k0) + ch1 * 16);
    async_copy16(lB + row1 * 112 + ch1 * 16,
                 (const char*)(B + (size_t)(colB + row1) * K + k0) + ch1 * 16);
  };

  v8f acc[4][2];
#pragma unroll
  for (int mt = 0; mt < 4; ++mt)
#pragma unroll
    for (int nt = 0; nt < 2; ++nt) acc[mt][nt] = (v8f){};

  const int ksteps = K >> 5;
  issue_stage(0, ldsA0, ldsB0);

  for (int ks = 0; ks < ksteps; ++ks) {
    const int buf = ks & 1;
    if (ks + 1 < ksteps) {
      issue_stage((ks + 1) << 5, buf ? ldsA0 : ldsA1, buf ? ldsB0 : ldsB1);
      wait_async_le4();
    } else {
      wait_async_0();
    }
    __syncthreads();

    const unsigned short* tA = &sAB[buf][0][0];
    const unsigned short* tB = &sAB[buf][1][0];
    union F { v16bf v; uint4 q[2]; };
    F fb[2];
#pragma unroll
    for (int nt = 0; nt < 2; ++nt) {
      int cc = (wn << 5) + (nt << 4) + l16;
      fb[nt].q[0] = *(const uint4*)(tB + cc * TSTRIDE + kbB);
      fb[nt].q[1] = *(const uint4*)(tB + cc * TSTRIDE + kbB + 16);
    }
#pragma unroll
    for (int mt = 0; mt < 4; ++mt) {
      int r = (wm << 6) + (mt << 4) + l16;
      F fa;
      fa.q[0] = *(const uint4*)(tA + r * TSTRIDE + kbA);
      fa.q[1] = *(const uint4*)(tA + r * TSTRIDE + kbA + 16);
#pragma unroll
      for (int nt = 0; nt < 2; ++nt)
        acc[mt][nt] = __builtin_amdgcn_wmma_f32_16x16x32_bf16(
            false, fa.v, false, fb[nt].v, (short)0, acc[mt][nt], false, false);
    }
    __syncthreads();
  }

#pragma unroll
  for (int mt = 0; mt < 4; ++mt) {
    int r0 = rowA + (wm << 6) + (mt << 4) + (half << 3);
#pragma unroll
    for (int nt = 0; nt < 2; ++nt) {
      int c0 = colB + (wn << 5) + (nt << 4) + l16;
      float* cp = C + (size_t)r0 * N + c0;
#pragma unroll
      for (int v = 0; v < 8; ++v) cp[(size_t)v * N] = acc[mt][nt][v];
    }
  }
}

// ---------------- SSD chunk kernel (WMMA Y_diag): one block per (b, chunk, head) ----
// LDS layout (bytes):
//   Cbf 256x72 bf16 (36,864) | Bbf 256x72 (36,864) | XT 32x264 (16,896)
//   Wst 8 waves x 16x56 (14,336) | sacum[256] | sw[256]          total 107,008 B
#define CSTR__  72
#define XTSTR__ 264
#define WSTR__  56
#define SSD_LDS_BYTES ((256*CSTR__ + 256*CSTR__ + 32*XTSTR__ + 8*16*WSTR__) * 2 + 2 * CS__ * 4)

__global__ void ssd_chunk_kernel(const float* __restrict__ xbcdt,
                                 const float* __restrict__ dt_bias,
                                 const float* __restrict__ A_log,
                                 float* __restrict__ acum_ws,
                                 float* __restrict__ states_ws,
                                 float* __restrict__ ydiag) {
  extern __shared__ char smem_raw[];
  unsigned short* Cbf = (unsigned short*)smem_raw;     // [256][CSTR]
  unsigned short* Bbf = Cbf + 256 * CSTR__;            // [256][CSTR]
  unsigned short* XT  = Bbf + 256 * CSTR__;            // [32][XTSTR]  xdt^T
  unsigned short* Wst = XT + 32 * XTSTR__;             // [8][16][WSTR]
  float* sacum = (float*)(Wst + 8 * 16 * WSTR__);
  float* sw    = sacum + CS__;

  const int blk = blockIdx.x;
  const int h = blk & (NH__ - 1);
  const int c = (blk >> 5) & (NC__ - 1);
  const int b = blk >> 9;
  const int i = threadIdx.x;
  const int lane = i & 31, w = i >> 5;
  const int half = lane >> 4, l16 = lane & 15;
  const int kbA = half << 3, kbB = half << 3;
  const int l = c * CS__ + i;
  const float* row = xbcdt + (size_t)(b * L__ + l) * XSTR__;

  float v   = row[DM__ + 2 * DS__ + h] + dt_bias[h];
  float dtv = (v > 20.f) ? v : log1pf(expf(v));       // softplus
  float Ah  = -expf(A_log[h]);
  sacum[i] = dtv * Ah;
  __syncthreads();
  for (int off = 1; off < CS__; off <<= 1) {          // inclusive cumsum
    float add = (i >= off) ? sacum[i - off] : 0.0f;
    __syncthreads();
    sacum[i] += add;
    __syncthreads();
  }
  float acum_i = sacum[i];
  float alast  = sacum[CS__ - 1];
  sw[i] = __expf(alast - acum_i);                     // state decay (dt folded into xdt)
  acum_ws[((size_t)(b * NC__ + c) * NH__ + h) * CS__ + i] = acum_i;

  // stage bf16 tiles: B rows, C rows, xdt transposed
#pragma unroll
  for (int k = 0; k < DS__; ++k) Bbf[i * CSTR__ + k] = f32_to_bf16(row[DM__ + k]);
#pragma unroll
  for (int k = 0; k < DS__; ++k) Cbf[i * CSTR__ + k] = f32_to_bf16(row[DM__ + DS__ + k]);
#pragma unroll
  for (int p = 0; p < HD__; ++p) XT[p * XTSTR__ + i] = f32_to_bf16(row[h * HD__ + p] * dtv);
  __syncthreads();

  unsigned short* WstW = Wst + w * 16 * WSTR__;       // per-wave staging tile
  union F { v16bf v; uint4 q[2]; };

  // each wave computes 2 row-tiles: ti = w and w+8
  for (int rep = 0; rep < 2; ++rep) {
    const int ti = w + (rep << 3);
    v8f yacc0 = (v8f){}, yacc1 = (v8f){};
    for (int tjp = 0; tjp <= (ti >> 1); ++tjp) {
      const int j0 = tjp << 5;
      // two 16x16 CB tiles (K = DS = 64), scale by L-mask, stage as bf16 A-fragments
#pragma unroll
      for (int sub = 0; sub < 2; ++sub) {
        const int tj = (tjp << 1) + sub;
        v8f cb = (v8f){};
#pragma unroll
        for (int k0 = 0; k0 < DS__; k0 += 32) {
          F fa, fb;
          const unsigned short* cr = Cbf + (ti * 16 + l16) * CSTR__ + k0;
          fa.q[0] = *(const uint4*)(cr + kbA);
          fa.q[1] = *(const uint4*)(cr + kbA + 16);
          const unsigned short* br = Bbf + (tj * 16 + l16) * CSTR__ + k0;
          fb.q[0] = *(const uint4*)(br + kbB);
          fb.q[1] = *(const uint4*)(br + kbB + 16);
          cb = __builtin_amdgcn_wmma_f32_16x16x32_bf16(
              false, fa.v, false, fb.v, (short)0, cb, false, false);
        }
        const int jj = tj * 16 + l16;
        const float aj = sacum[jj];
#pragma unroll
        for (int e = 0; e < 8; ++e) {
          const int ii = ti * 16 + e + (half << 3);
          float wv = cb[e] * __expf(sacum[ii] - aj);
          wv = (jj <= ii) ? wv : 0.f;                 // causal mask (handles diagonal)
          WstW[(e + (half << 3)) * WSTR__ + (sub << 4) + l16] = f32_to_bf16(wv);
        }
      }
      // accumulate Y[i,p] += W(16x32) @ xdt(32x32-slice): 2 WMMAs (p tiles 0,1)
      F wa, xb0, xb1;
      wa.q[0] = *(const uint4*)(WstW + l16 * WSTR__ + kbA);
      wa.q[1] = *(const uint4*)(WstW + l16 * WSTR__ + kbA + 16);
      xb0.q[0] = *(const uint4*)(XT + l16 * XTSTR__ + j0 + kbB);
      xb0.q[1] = *(const uint4*)(XT + l16 * XTSTR__ + j0 + kbB + 16);
      xb1.q[0] = *(const uint4*)(XT + (16 + l16) * XTSTR__ + j0 + kbB);
      xb1.q[1] = *(const uint4*)(XT + (16 + l16) * XTSTR__ + j0 + kbB + 16);
      yacc0 = __builtin_amdgcn_wmma_f32_16x16x32_bf16(
          false, wa.v, false, xb0.v, (short)0, yacc0, false, false);
      yacc1 = __builtin_amdgcn_wmma_f32_16x16x32_bf16(
          false, wa.v, false, xb1.v, (short)0, yacc1, false, false);
    }
    // store Y tile: rows ti*16 + e + 8*half, cols h*32 + {l16, 16+l16}
    float* yo = ydiag + (size_t)(b * L__ + c * CS__ + ti * 16 + (half << 3)) * DM__
              + h * HD__ + l16;
#pragma unroll
    for (int e = 0; e < 8; ++e) {
      yo[(size_t)e * DM__]      = yacc0[e];
      yo[(size_t)e * DM__ + 16] = yacc1[e];
    }
  }

  // chunk states[p,n] = sum_j B[j,n] * exp(alast-acum_j) * xdt[j,p]   (VALU, small)
  const int p  = i & (HD__ - 1);
  const int nb = (i >> 5) * 8;
  float st[8];
#pragma unroll
  for (int k = 0; k < 8; ++k) st[k] = 0.f;
  for (int j = 0; j < CS__; ++j) {
    float wx = sw[j] * bf16_to_f32(XT[p * XTSTR__ + j]);
#pragma unroll
    for (int k = 0; k < 8; ++k) st[k] += bf16_to_f32(Bbf[j * CSTR__ + nb + k]) * wx;
  }
  float* sp = states_ws + (((size_t)(b * NC__ + c) * NH__ + h) * HD__ + p) * DS__ + nb;
#pragma unroll
  for (int k = 0; k < 8; ++k) sp[k] = st[k];
}

// ---------------- inter-chunk scan: one block per (b, head) ----------------
__global__ void scan_kernel(const float* __restrict__ states_ws,
                            const float* __restrict__ acum_ws,
                            float* __restrict__ prev_ws) {
  int bh = blockIdx.x;
  int h = bh & (NH__ - 1);
  int b = bh >> 5;
  int t = threadIdx.x;
  float s[8];
#pragma unroll
  for (int k = 0; k < 8; ++k) s[k] = 0.f;
  for (int c = 0; c < NC__; ++c) {
    size_t idx = ((size_t)(b * NC__ + c) * NH__ + h);
    float dec = __expf(acum_ws[idx * CS__ + (CS__ - 1)]);
    size_t base = idx * (HD__ * DS__) + (size_t)t * 8;
#pragma unroll
    for (int k = 0; k < 8; ++k) {
      prev_ws[base + k] = s[k];
      s[k] = s[k] * dec + states_ws[base + k];
    }
  }
}

// ---------------- Y_off + skip term, emit bf16 ----------------
__global__ void yoff_kernel(const float* __restrict__ xbcdt,
                            const float* __restrict__ acum_ws,
                            const float* __restrict__ prev_ws,
                            const float* __restrict__ ydiag,
                            const float* __restrict__ Dvec,
                            unsigned short* __restrict__ ybf) {
  __shared__ float sprev[HD__ * DS__];
  int blk = blockIdx.x;
  int h = blk & (NH__ - 1);
  int c = (blk >> 5) & (NC__ - 1);
  int b = blk >> 9;
  int i = threadIdx.x;
  size_t pbase = ((size_t)(b * NC__ + c) * NH__ + h) * (HD__ * DS__);
#pragma unroll
  for (int k = 0; k < 8; ++k) sprev[i * 8 + k] = prev_ws[pbase + i * 8 + k];
  __syncthreads();

  int l = c * CS__ + i;
  const float* row = xbcdt + (size_t)(b * L__ + l) * XSTR__;
  float creg[DS__];
#pragma unroll
  for (int n = 0; n < DS__; ++n) creg[n] = row[DM__ + DS__ + n];
  float ea = __expf(acum_ws[((size_t)(b * NC__ + c) * NH__ + h) * CS__ + i]);
  float Dh = Dvec[h];
  const float* yd = ydiag + (size_t)(b * L__ + l) * DM__ + h * HD__;
  unsigned short* yo = ybf + (size_t)(b * L__ + l) * DM__ + h * HD__;
  for (int p = 0; p < HD__; ++p) {
    float acc = 0.f;
#pragma unroll
    for (int n = 0; n < DS__; ++n) acc += creg[n] * sprev[p * DS__ + n];
    float val = yd[p] + ea * acc + row[h * HD__ + p] * Dh;
    yo[p] = f32_to_bf16(val);
  }
}

// ---------------- host launcher ----------------
extern "C" void kernel_launch(void* const* d_in, const int* in_sizes, int n_in,
                              void* d_out, int out_size, void* d_ws, size_t ws_size,
                              hipStream_t stream) {
  const float* u       = (const float*)d_in[0];
  const float* W_in    = (const float*)d_in[1];
  const float* dt_bias = (const float*)d_in[2];
  const float* A_log   = (const float*)d_in[3];
  const float* Dvec    = (const float*)d_in[4];
  const float* W_out   = (const float*)d_in[5];
  float* out = (float*)d_out;
  (void)in_sizes; (void)n_in; (void)out_size; (void)ws_size;

  const size_t M = (size_t)B__ * L__;   // 8192 rows

  char* wp = (char*)d_ws;
  auto carve = [&](size_t bytes) -> char* {
    char* p = wp; wp += (bytes + 255) & ~(size_t)255; return p;
  };
  unsigned short* ubf   = (unsigned short*)carve(M * DM__ * 2);
  unsigned short* wibf  = (unsigned short*)carve((size_t)XSTR__ * DM__ * 2);
  unsigned short* wobf  = (unsigned short*)carve((size_t)DM__ * DM__ * 2);
  float* xbcdt  = (float*)carve(M * XSTR__ * 4);
  float* acum   = (float*)carve((size_t)B__ * NC__ * NH__ * CS__ * 4);
  float* states = (float*)carve((size_t)B__ * NC__ * NH__ * HD__ * DS__ * 4);
  float* prev   = (float*)carve((size_t)B__ * NC__ * NH__ * HD__ * DS__ * 4);
  float* ydiag  = (float*)carve(M * DM__ * 4);
  unsigned short* ybf = (unsigned short*)carve(M * DM__ * 2);

  // 1) fp32 -> bf16 casts (+ zero the padded weight rows first)
  hipMemsetAsync(wibf, 0, (size_t)XSTR__ * DM__ * 2, stream);
  {
    int n = (int)(M * DM__);
    cast_f32_bf16<<<(n + 255) / 256, 256, 0, stream>>>(u, ubf, n);
    n = DPROJ__ * DM__;
    cast_f32_bf16<<<(n + 255) / 256, 256, 0, stream>>>(W_in, wibf, n);
    n = DM__ * DM__;
    cast_f32_bf16<<<(n + 255) / 256, 256, 0, stream>>>(W_out, wobf, n);
  }
  // 2) in-projection GEMM (async-LDS double-buffered WMMA), N padded to 1280
  {
    dim3 grid(XSTR__ / 128, (int)(M / 128));
    gemm_bf16_wmma<<<grid, 256, 0, stream>>>(ubf, wibf, xbcdt, (int)M, XSTR__, DM__);
  }
  // 3) SSD chunk kernel: WMMA Y_diag (107KB dynamic LDS; CDNA5 WGP has 320KB)
  {
    hipFuncSetAttribute((const void*)ssd_chunk_kernel,
                        hipFuncAttributeMaxDynamicSharedMemorySize, SSD_LDS_BYTES);
    ssd_chunk_kernel<<<B__ * NC__ * NH__, CS__, SSD_LDS_BYTES, stream>>>(
        xbcdt, dt_bias, A_log, acum, states, ydiag);
  }
  // 4) inter-chunk scan
  scan_kernel<<<B__ * NH__, 256, 0, stream>>>(states, acum, prev);
  // 5) off-diagonal + skip, emit bf16 activations
  yoff_kernel<<<B__ * NC__ * NH__, CS__, 0, stream>>>(xbcdt, acum, prev, ydiag,
                                                      Dvec, ybf);
  // 6) out-projection GEMM -> d_out fp32
  {
    dim3 grid(DM__ / 128, (int)(M / 128));
    gemm_bf16_wmma<<<grid, 256, 0, stream>>>(ybf, wobf, out, (int)M, DM__, DM__);
  }
}
